// HierarchicalDifferentialAttention_28501402976987
// MI455X (gfx1250) — matrix operand order repaired and checked by reading, not compiled, not run on hardware
//
#include <hip/hip_runtime.h>
#include <hip/hip_bf16.h>
#include <math.h>

// Problem constants
#define Bn 4
#define Sn 1024
#define Dn 1024
#define Hn 16
#define DHn 64

typedef __attribute__((ext_vector_type(16))) __bf16 v16bf;
typedef __attribute__((ext_vector_type(8)))  float  v8f;

union Frag {
    uint4 u[2];
    v16bf v;
};

static __device__ __forceinline__ unsigned short f2bf(float f) {
    unsigned int u = __float_as_uint(f);
    u += 0x7FFFu + ((u >> 16) & 1u);     // round-to-nearest-even
    return (unsigned short)(u >> 16);
}

// CDNA5 async global->LDS copy (16 bytes per lane), GVS addressing:
// dsaddr = lds (per-lane), mem = SGPR base + per-lane byte offset.
static __device__ __forceinline__ void async_copy16(unsigned lds, const void* sbase,
                                                    unsigned voff) {
    asm volatile("global_load_async_to_lds_b128 %0, %1, %2"
                 :: "v"(lds), "v"(voff), "s"(sbase)
                 : "memory");
}
template <int N>
static __device__ __forceinline__ void async_waitN() {
#if __has_builtin(__builtin_amdgcn_s_wait_asynccnt)
    __builtin_amdgcn_s_wait_asynccnt(N);
#else
    asm volatile("s_wait_asynccnt %0" :: "i"(N) : "memory");
#endif
}
static __device__ __forceinline__ unsigned lds_addr(const void* p) {
    return (unsigned)(unsigned long long)p;   // low 32 bits of flat shared addr = LDS addr
}

// ---------------------------------------------------------------------------
// f32 -> bf16 bulk convert (float4 -> ushort4 per thread)
// ---------------------------------------------------------------------------
__global__ __launch_bounds__(256)
void cvt_bf16_kernel(const float* __restrict__ src, unsigned short* __restrict__ dst,
                     int n4) {
    int i = blockIdx.x * 256 + threadIdx.x;
    if (i >= n4) return;
    float4 v = ((const float4*)src)[i];
    ushort4 o;
    o.x = f2bf(v.x); o.y = f2bf(v.y); o.z = f2bf(v.z); o.w = f2bf(v.w);
    ((ushort4*)dst)[i] = o;
}

// ---------------------------------------------------------------------------
// GEMM:  C[M][N] = A[M][K] @ W[N][K]^T + bias[N]  (A, W already bf16)
// 256 threads / 8 waves; tile 64(M) x 128(N); K chunked by 32.
// Double-buffered async global->LDS staging; 4 WMMAs per wave per chunk.
// ---------------------------------------------------------------------------
__global__ __launch_bounds__(256)
void gemm_bf16_wmma(const unsigned short* __restrict__ A,
                    const unsigned short* __restrict__ W,
                    const float* __restrict__ bias, float* __restrict__ C,
                    int M, int N, int K) {
    __shared__ unsigned short At[2][64][32];
    __shared__ unsigned short Bt[2][128][32];

    const int tid  = threadIdx.x;
    const int lane = tid & 31, wid = tid >> 5;
    const int half = lane >> 4, l16 = lane & 15;
    const int wm = wid & 3, wn = wid >> 2;          // wn: 0..1 (64 cols each)
    const int n0 = blockIdx.x * 128, m0 = blockIdx.y * 64;

    const int arow = tid >> 2, aseg = (tid & 3) * 8;   // A: 64 rows, 1 x b128/thread
    const int brow = tid >> 1, bseg = (tid & 1) * 16;  // B: 128 rows, 2 x b128/thread

    const unsigned ldsA0  = lds_addr(&At[0][arow][aseg]);
    const unsigned ldsA1  = lds_addr(&At[1][arow][aseg]);
    const unsigned ldsB00 = lds_addr(&Bt[0][brow][bseg]);
    const unsigned ldsB10 = lds_addr(&Bt[1][brow][bseg]);
    const unsigned voffA  = (unsigned)(arow * K + aseg) * 2;
    const unsigned voffB0 = (unsigned)(brow * K + bseg) * 2;
    const unsigned voffB1 = voffB0 + 16;

    const unsigned short* Abase = A + (size_t)m0 * K;
    const unsigned short* Wbase = W + (size_t)n0 * K;

    v8f acc[4] = {{}, {}, {}, {}};

    // prologue: issue chunk 0 into buffer 0
    async_copy16(ldsA0,       Abase, voffA);
    async_copy16(ldsB00,      Wbase, voffB0);
    async_copy16(ldsB00 + 16, Wbase, voffB1);

    int cur = 0;
    for (int k0 = 0; k0 < K; k0 += 32) {
        if (k0 + 32 < K) {  // issue next chunk into the other buffer
            unsigned la = cur ? ldsA0 : ldsA1;
            unsigned lb = cur ? ldsB00 : ldsB10;
            async_copy16(la,      Abase + k0 + 32, voffA);
            async_copy16(lb,      Wbase + k0 + 32, voffB0);
            async_copy16(lb + 16, Wbase + k0 + 32, voffB1);
            async_waitN<3>();          // previous chunk's 3 ops complete (in-order)
        } else {
            async_waitN<0>();
        }
        __syncthreads();

        Frag fa, fb[4];
        fa.u[0] = *(const uint4*)&At[cur][wm*16 + l16][8*half];
        fa.u[1] = *(const uint4*)&At[cur][wm*16 + l16][16 + 8*half];
#pragma unroll
        for (int t = 0; t < 4; ++t) {
            fb[t].u[0] = *(const uint4*)&Bt[cur][wn*64 + t*16 + l16][8*half];
            fb[t].u[1] = *(const uint4*)&Bt[cur][wn*64 + t*16 + l16][16 + 8*half];
        }
#pragma unroll
        for (int t = 0; t < 4; ++t)
            acc[t] = __builtin_amdgcn_wmma_f32_16x16x32_bf16(false, fa.v, false, fb[t].v,
                                                             (short)0, acc[t], false, false);
        __syncthreads();
        cur ^= 1;
    }

#pragma unroll
    for (int t = 0; t < 4; ++t)
        for (int r = 0; r < 8; ++r) {
            int m = m0 + wm*16 + r + 8*half;
            int n = n0 + wn*64 + t*16 + l16;
            C[(size_t)m * N + n] = acc[t][r] + bias[n];
        }
}

// ---------------------------------------------------------------------------
// RoPE: q_proj/k_proj (B,S,2D f32) -> rotated per-head bf16 (B,H,S,Dh)
// ---------------------------------------------------------------------------
__global__ __launch_bounds__(256)
void rope_kernel(const float* __restrict__ qproj, const float* __restrict__ kproj,
                 unsigned short* __restrict__ q1b, unsigned short* __restrict__ q2b,
                 unsigned short* __restrict__ k1b, unsigned short* __restrict__ k2b) {
    int tid = blockIdx.x * 256 + threadIdx.x;     // < B*H*S*32
    int j  = tid & 31;
    int s  = (tid >> 5) & (Sn - 1);
    int bh = tid >> 15;
    int b  = bh >> 4, h = bh & 15;

    float freq = __powf(10000.0f, -(float)j / 32.0f);
    float ang  = (float)s * freq;
    float sn, cs;
    __sincosf(ang, &sn, &cs);

    size_t rowq = ((size_t)b * Sn + s) * (2 * Dn);
    size_t ro   = ((size_t)bh * Sn + s) * DHn + j;
    int c1 = h * DHn + j;

    {   float x1 = qproj[rowq + c1], x2 = qproj[rowq + c1 + 32];
        q1b[ro] = f2bf(x1*cs - x2*sn);  q1b[ro + 32] = f2bf(x1*sn + x2*cs); }
    {   float x1 = qproj[rowq + Dn + c1], x2 = qproj[rowq + Dn + c1 + 32];
        q2b[ro] = f2bf(x1*cs - x2*sn);  q2b[ro + 32] = f2bf(x1*sn + x2*cs); }
    {   float x1 = kproj[rowq + c1], x2 = kproj[rowq + c1 + 32];
        k1b[ro] = f2bf(x1*cs - x2*sn);  k1b[ro + 32] = f2bf(x1*sn + x2*cs); }
    {   float x1 = kproj[rowq + Dn + c1], x2 = kproj[rowq + Dn + c1 + 32];
        k2b[ro] = f2bf(x1*cs - x2*sn);  k2b[ro + 32] = f2bf(x1*sn + x2*cs); }
}

// ---------------------------------------------------------------------------
// V prep: vproj (B,S,D f32) -> per-head transposed bf16 vtr[(b,h,d), s]
// ---------------------------------------------------------------------------
__global__ __launch_bounds__(256)
void vprep_kernel(const float* __restrict__ vproj, unsigned short* __restrict__ vtr) {
    int i = blockIdx.x * 256 + threadIdx.x;       // over (b,s,c), coalesced read
    int c = i & (Dn - 1);
    int s = (i >> 10) & (Sn - 1);
    int b = i >> 20;
    int h = c >> 6, d = c & 63;
    vtr[(((size_t)(b * Hn + h) * DHn + d) << 10) + s] = f2bf(vproj[i]);
}

// ---------------------------------------------------------------------------
// Scores: raw[b,h,q,k] = (Q.K)/8 + padding mask; into a1/a2 slots.
// grid = (S/128 key tiles, S/64 query tiles, B*H*2); K-loop = Dh in 2 chunks,
// double-buffered async staging.
// ---------------------------------------------------------------------------
__global__ __launch_bounds__(256)
void scores_kernel(const unsigned short* __restrict__ q1, const unsigned short* __restrict__ q2,
                   const unsigned short* __restrict__ k1, const unsigned short* __restrict__ k2,
                   const int* __restrict__ pad,
                   float* __restrict__ a1, float* __restrict__ a2) {
    __shared__ unsigned short Qt[2][64][32];
    __shared__ unsigned short Kt[2][128][32];

    const int z = blockIdx.z;
    const int branch = z & 1, bh = z >> 1;
    const int b = bh >> 4;
    const unsigned short* Qs = branch ? q2 : q1;
    const unsigned short* Ks = branch ? k2 : k1;
    float* O = branch ? a2 : a1;

    const int tid  = threadIdx.x;
    const int lane = tid & 31, wid = tid >> 5;
    const int half = lane >> 4, l16 = lane & 15;
    const int wm = wid & 3, wn = wid >> 2;
    const int n0 = blockIdx.x * 128, q0 = blockIdx.y * 64;

    const int arow = tid >> 2, aseg = (tid & 3) * 8;
    const int brow = tid >> 1, bseg = (tid & 1) * 16;
    const unsigned ldsA0  = lds_addr(&Qt[0][arow][aseg]);
    const unsigned ldsA1  = lds_addr(&Qt[1][arow][aseg]);
    const unsigned ldsB00 = lds_addr(&Kt[0][brow][bseg]);
    const unsigned ldsB10 = lds_addr(&Kt[1][brow][bseg]);
    const unsigned voffA  = (unsigned)(arow * DHn + aseg) * 2;
    const unsigned voffB0 = (unsigned)(brow * DHn + bseg) * 2;
    const unsigned voffB1 = voffB0 + 16;

    const unsigned short* Qbase = Qs + ((size_t)bh * Sn + q0) * DHn;
    const unsigned short* Kbase = Ks + ((size_t)bh * Sn + n0) * DHn;

    v8f acc[4] = {{}, {}, {}, {}};

    async_copy16(ldsA0,       Qbase, voffA);
    async_copy16(ldsB00,      Kbase, voffB0);
    async_copy16(ldsB00 + 16, Kbase, voffB1);

    int cur = 0;
    for (int k0 = 0; k0 < DHn; k0 += 32) {
        if (k0 + 32 < DHn) {
            unsigned la = cur ? ldsA0 : ldsA1;
            unsigned lb = cur ? ldsB00 : ldsB10;
            async_copy16(la,      Qbase + k0 + 32, voffA);
            async_copy16(lb,      Kbase + k0 + 32, voffB0);
            async_copy16(lb + 16, Kbase + k0 + 32, voffB1);
            async_waitN<3>();
        } else {
            async_waitN<0>();
        }
        __syncthreads();

        Frag fa, fb[4];
        fa.u[0] = *(const uint4*)&Qt[cur][wm*16 + l16][8*half];
        fa.u[1] = *(const uint4*)&Qt[cur][wm*16 + l16][16 + 8*half];
#pragma unroll
        for (int t = 0; t < 4; ++t) {
            fb[t].u[0] = *(const uint4*)&Kt[cur][wn*64 + t*16 + l16][8*half];
            fb[t].u[1] = *(const uint4*)&Kt[cur][wn*64 + t*16 + l16][16 + 8*half];
        }
#pragma unroll
        for (int t = 0; t < 4; ++t)
            acc[t] = __builtin_amdgcn_wmma_f32_16x16x32_bf16(false, fa.v, false, fb[t].v,
                                                             (short)0, acc[t], false, false);
        __syncthreads();
        cur ^= 1;
    }

#pragma unroll
    for (int t = 0; t < 4; ++t)
        for (int r = 0; r < 8; ++r) {
            int qi = q0 + wm*16 + r + 8*half;
            int kc = n0 + wn*64 + t*16 + l16;
            float v = acc[t][r] * 0.125f;            // 1/sqrt(64)
            if (pad[b * Sn + kc] == 0) v = -1e9f;
            O[((size_t)bh * Sn + qi) * Sn + kc] = v;
        }
}

// ---------------------------------------------------------------------------
// Row-wise softmax in place; grid = (B*H*S, 2), 256 threads x 4 elements.
// ---------------------------------------------------------------------------
__global__ __launch_bounds__(256)
void softmax_kernel(float* __restrict__ a1, float* __restrict__ a2) {
    float* rowp = (blockIdx.y ? a2 : a1) + (size_t)blockIdx.x * Sn;
    const int tid = threadIdx.x;
    __shared__ float red[256];

    float4 v = *(float4*)(rowp + tid * 4);
    float m = fmaxf(fmaxf(v.x, v.y), fmaxf(v.z, v.w));
    red[tid] = m; __syncthreads();
    for (int s = 128; s > 0; s >>= 1) {
        if (tid < s) red[tid] = fmaxf(red[tid], red[tid + s]);
        __syncthreads();
    }
    m = red[0]; __syncthreads();

    v.x = __expf(v.x - m); v.y = __expf(v.y - m);
    v.z = __expf(v.z - m); v.w = __expf(v.w - m);
    red[tid] = v.x + v.y + v.z + v.w; __syncthreads();
    for (int s = 128; s > 0; s >>= 1) {
        if (tid < s) red[tid] += red[tid + s];
        __syncthreads();
    }
    float inv = 1.0f / red[0];
    v.x *= inv; v.y *= inv; v.z *= inv; v.w *= inv;
    *(float4*)(rowp + tid * 4) = v;
}

// ---------------------------------------------------------------------------
// Context: combined = C ⊙ (a1 - λ a2) built tile-wise (f32->bf16 in LDS);
// V pre-transposed bf16, double-buffered async copy. merged = combined @ V.
// grid = (S/64 query tiles, B*H). N = Dh = 64 per block.
// ---------------------------------------------------------------------------
__global__ __launch_bounds__(256)
void context_kernel(const float* __restrict__ a1, const float* __restrict__ a2,
                    const float* __restrict__ maskC, const float* __restrict__ lam_p,
                    const unsigned short* __restrict__ vtr, float* __restrict__ merged) {
    __shared__ unsigned short Ct[64][32];
    __shared__ unsigned short Vt[2][64][32];

    const int bh = blockIdx.y;
    const int b = bh >> 4, h = bh & 15;
    const int q0 = blockIdx.x * 64;
    const float lam = *lam_p;

    const int tid  = threadIdx.x;
    const int lane = tid & 31, wid = tid >> 5;
    const int half = lane >> 4, l16 = lane & 15;
    const int wm = wid & 3, wn = wid >> 2;
    const int row = tid >> 2, cg = (tid & 3) * 8;
    const unsigned ldsV0 = lds_addr(&Vt[0][row][cg]);
    const unsigned ldsV1 = lds_addr(&Vt[1][row][cg]);
    const unsigned voffV = (unsigned)(row * Sn + cg) * 2;

    const unsigned short* Vbase = vtr + ((size_t)bh * DHn) * Sn;

    v8f acc0 = {}; v8f acc1 = {};

    async_copy16(ldsV0, Vbase, voffV);

    int cur = 0;
    for (int k0 = 0; k0 < Sn; k0 += 32) {
        if (k0 + 32 < Sn)
            async_copy16(cur ? ldsV0 : ldsV1, Vbase + k0 + 32, voffV);

        size_t abase = ((size_t)bh * Sn + (q0 + row)) * Sn + k0 + cg;
        size_t mbase = (size_t)(q0 + row) * Sn + k0 + cg;
        float4 p1a = *(const float4*)(a1 + abase);
        float4 p1b = *(const float4*)(a1 + abase + 4);
        float4 p2a = *(const float4*)(a2 + abase);
        float4 p2b = *(const float4*)(a2 + abase + 4);
        float4 ma  = *(const float4*)(maskC + mbase);
        float4 mb  = *(const float4*)(maskC + mbase + 4);
        if (k0 + 32 < Sn) {
            __builtin_prefetch(a1 + abase + 32, 0, 1);
            __builtin_prefetch(a2 + abase + 32, 0, 1);
        }
        Ct[row][cg+0]=f2bf(ma.x*(p1a.x - lam*p2a.x));
        Ct[row][cg+1]=f2bf(ma.y*(p1a.y - lam*p2a.y));
        Ct[row][cg+2]=f2bf(ma.z*(p1a.z - lam*p2a.z));
        Ct[row][cg+3]=f2bf(ma.w*(p1a.w - lam*p2a.w));
        Ct[row][cg+4]=f2bf(mb.x*(p1b.x - lam*p2b.x));
        Ct[row][cg+5]=f2bf(mb.y*(p1b.y - lam*p2b.y));
        Ct[row][cg+6]=f2bf(mb.z*(p1b.z - lam*p2b.z));
        Ct[row][cg+7]=f2bf(mb.w*(p1b.w - lam*p2b.w));

        if (k0 + 32 < Sn) async_waitN<1>();   // current V chunk done (in-order)
        else              async_waitN<0>();
        __syncthreads();

        Frag fa, fb0, fb1;
        fa.u[0]  = *(const uint4*)&Ct[wm*16 + l16][8*half];
        fa.u[1]  = *(const uint4*)&Ct[wm*16 + l16][16 + 8*half];
        fb0.u[0] = *(const uint4*)&Vt[cur][wn*32 + l16][8*half];
        fb0.u[1] = *(const uint4*)&Vt[cur][wn*32 + l16][16 + 8*half];
        fb1.u[0] = *(const uint4*)&Vt[cur][wn*32 + 16 + l16][8*half];
        fb1.u[1] = *(const uint4*)&Vt[cur][wn*32 + 16 + l16][16 + 8*half];

        acc0 = __builtin_amdgcn_wmma_f32_16x16x32_bf16(false, fa.v, false, fb0.v,
                                                       (short)0, acc0, false, false);
        acc1 = __builtin_amdgcn_wmma_f32_16x16x32_bf16(false, fa.v, false, fb1.v,
                                                       (short)0, acc1, false, false);
        __syncthreads();
        cur ^= 1;
    }

    for (int r = 0; r < 8; ++r) {
        int qi = q0 + wm*16 + r + 8*half;
        int da = wn*32 + l16;
        int db = da + 16;
        merged[((size_t)b * Sn + qi) * Dn + h * DHn + da] = acc0[r];
        merged[((size_t)b * Sn + qi) * Dn + h * DHn + db] = acc1[r];
    }
}

// ---------------------------------------------------------------------------
// GroupNorm over (S,Dh) per (b,h) + gamma/beta + (1-LAMBDA_INIT); writes bf16.
// ---------------------------------------------------------------------------
__global__ __launch_bounds__(256)
void groupnorm_kernel(const float* __restrict__ merged,
                      const float* __restrict__ gn_w, const float* __restrict__ gn_b,
                      unsigned short* __restrict__ scaledb) {
    const int bh = blockIdx.x;
    const int b = bh >> 4, h = bh & 15;
    const int tid = threadIdx.x;
    __shared__ float s1[256], s2[256];

    float sum = 0.f, sq = 0.f;
    for (int i = tid; i < Sn * DHn; i += 256) {
        int s = i >> 6, d = i & 63;
        float v = merged[((size_t)b * Sn + s) * Dn + h * DHn + d];
        sum += v; sq += v * v;
    }
    s1[tid] = sum; s2[tid] = sq; __syncthreads();
    for (int st = 128; st > 0; st >>= 1) {
        if (tid < st) { s1[tid] += s1[tid + st]; s2[tid] += s2[tid + st]; }
        __syncthreads();
    }
    float mean = s1[0] * (1.0f / 65536.0f);
    float var  = s2[0] * (1.0f / 65536.0f) - mean * mean;
    float rstd = rsqrtf(var + 1e-5f);
    __syncthreads();

    for (int i = tid; i < Sn * DHn; i += 256) {
        int s = i >> 6, d = i & 63;
        int c = h * DHn + d;
        size_t idx = ((size_t)b * Sn + s) * Dn + c;
        float v = merged[idx];
        scaledb[idx] = f2bf(((v - mean) * rstd * gn_w[c] + gn_b[c]) * 0.2f);
    }
}

// ---------------------------------------------------------------------------
extern "C" void kernel_launch(void* const* d_in, const int* in_sizes, int n_in,
                              void* d_out, int out_size, void* d_ws, size_t ws_size,
                              hipStream_t stream) {
    const float* x_norm = (const float*)d_in[0];
    const float* lam    = (const float*)d_in[1];
    const float* maskC  = (const float*)d_in[2];
    const int*   pad    = (const int*)d_in[3];
    const float* wq_w   = (const float*)d_in[4];
    const float* wq_b   = (const float*)d_in[5];
    const float* wk_w   = (const float*)d_in[6];
    const float* wk_b   = (const float*)d_in[7];
    const float* wv_w   = (const float*)d_in[8];
    const float* wv_b   = (const float*)d_in[9];
    const float* out_w  = (const float*)d_in[10];
    const float* out_b  = (const float*)d_in[11];
    const float* gn_w   = (const float*)d_in[12];
    const float* gn_b   = (const float*)d_in[13];

    float* out = (float*)d_out;                    // (B,S,D)
    float* a1  = out + (size_t)Bn * Sn * Dn;       // (B,H,S,S)
    float* a2  = a1 + (size_t)Bn * Hn * Sn * Sn;

    char* wsb = (char*)d_ws;
    float* qproj           = (float*)(wsb + 0);                    // 32 MB (B,S,2D)
    float* kproj           = (float*)(wsb + 33554432ull);          // 32 MB
    float* vproj           = (float*)(wsb + 67108864ull);          // 16 MB (B,S,D)
    unsigned short* q1b    = (unsigned short*)(wsb + 83886080ull); // 8 MB (B,H,S,Dh)
    unsigned short* q2b    = (unsigned short*)(wsb + 92274688ull);
    unsigned short* k1b    = (unsigned short*)(wsb + 100663296ull);
    unsigned short* k2b    = (unsigned short*)(wsb + 109051904ull);
    unsigned short* vtr    = (unsigned short*)(wsb + 117440512ull); // 8 MB (B,H,Dh,S)
    unsigned short* xb     = (unsigned short*)(wsb + 125829120ull); // 8 MB
    unsigned short* wqb    = (unsigned short*)(wsb + 134217728ull); // 4 MB
    unsigned short* wkb    = (unsigned short*)(wsb + 138412032ull); // 4 MB
    unsigned short* wvb    = (unsigned short*)(wsb + 142606336ull); // 2 MB
    unsigned short* owb    = (unsigned short*)(wsb + 144703488ull); // 2 MB
    float* merged          = (float*)(wsb + 0);                    // reuse qproj
    unsigned short* scaledb= (unsigned short*)(wsb + 33554432ull); // reuse kproj

    const int M = Bn * Sn;             // 4096

    // 0: convert operands to bf16
    cvt_bf16_kernel<<<4096, 256, 0, stream>>>(x_norm, xb, (Bn*Sn*Dn)/4);
    cvt_bf16_kernel<<<2048, 256, 0, stream>>>(wq_w, wqb, (2*Dn*Dn)/4);
    cvt_bf16_kernel<<<2048, 256, 0, stream>>>(wk_w, wkb, (2*Dn*Dn)/4);
    cvt_bf16_kernel<<<1024, 256, 0, stream>>>(wv_w, wvb, (Dn*Dn)/4);
    cvt_bf16_kernel<<<1024, 256, 0, stream>>>(out_w, owb, (Dn*Dn)/4);

    // 1-3: projections (N tiles of 128)
    gemm_bf16_wmma<<<dim3(2048/128, M/64), 256, 0, stream>>>(xb, wqb, wq_b, qproj, M, 2048, Dn);
    gemm_bf16_wmma<<<dim3(2048/128, M/64), 256, 0, stream>>>(xb, wkb, wk_b, kproj, M, 2048, Dn);
    gemm_bf16_wmma<<<dim3(1024/128, M/64), 256, 0, stream>>>(xb, wvb, wv_b, vproj, M, 1024, Dn);

    // 4: RoPE + head split (writes bf16); 5: V transpose/convert
    rope_kernel<<<(Bn*Hn*Sn*32)/256, 256, 0, stream>>>(qproj, kproj, q1b, q2b, k1b, k2b);
    vprep_kernel<<<(Bn*Sn*Dn)/256, 256, 0, stream>>>(vproj, vtr);

    // 6: raw scores into a1/a2 slots
    scores_kernel<<<dim3(Sn/128, Sn/64, Bn*Hn*2), 256, 0, stream>>>(q1b, q2b, k1b, k2b, pad, a1, a2);

    // 7: softmax in place
    softmax_kernel<<<dim3(Bn*Hn*Sn, 2), 256, 0, stream>>>(a1, a2);

    // 8: combined @ V -> merged (f32)
    context_kernel<<<dim3(Sn/64, Bn*Hn), 256, 0, stream>>>(a1, a2, maskC, lam, vtr, merged);

    // 9: group norm + scale -> bf16
    groupnorm_kernel<<<Bn*Hn, 256, 0, stream>>>(merged, gn_w, gn_b, scaledb);

    // 10: output projection
    gemm_bf16_wmma<<<dim3(1024/128, M/64), 256, 0, stream>>>(scaledb, owb, out_b, out, M, 1024, Dn);
}
